// MultiHeadAttention_68667937128487
// MI455X (gfx1250) — matrix operand order repaired and checked by reading, not compiled, not run on hardware
//
#include <hip/hip_runtime.h>

// ---- plain ext-vector types (avoid HIP struct-vector unions) ----
typedef __attribute__((ext_vector_type(16))) __bf16        v16bf;
typedef __attribute__((ext_vector_type(8)))  float         v8f;
typedef __attribute__((ext_vector_type(4)))  unsigned int  u32x4;
typedef __attribute__((ext_vector_type(4)))  float         f32x4;
typedef __attribute__((ext_vector_type(2)))  float         f32x2;

constexpr int BATCH = 4;
constexpr int T     = 2048;
constexpr int U     = 1024;
constexpr int H     = 16;
constexpr int D     = 64;
constexpr int KP    = 72;   // LDS row pitch in bf16 elems (rows 16B aligned)
// fold 1/sqrt(64) AND log2(e) into Q so scores are already in exp2 domain
constexpr float QSC = 0.125f * 1.4426950408889634f;

static __device__ __forceinline__ unsigned int pack_bf2(float x, float y) {
  unsigned short lo = __builtin_bit_cast(unsigned short, (__bf16)x);
  unsigned short hi = __builtin_bit_cast(unsigned short, (__bf16)y);
  return (unsigned int)lo | ((unsigned int)hi << 16);
}

static __device__ __forceinline__ v16bf load_bf16x16(const __bf16* p) {
  union { u32x4 q[2]; v16bf v; } u;
  u.q[0] = *(const u32x4*)(p);
  u.q[1] = *(const u32x4*)(p + 8);
  return u.v;
}

static __device__ __forceinline__ v8f wmma_bf16(v16bf a, v16bf b, v8f c) {
  // D = A(16x32) * B(32x16) + C, f32 accumulate
  return __builtin_amdgcn_wmma_f32_16x16x32_bf16(false, a, false, b, (short)0, c, false, false);
}

// raw v_max_num_f32 (no canonicalize self-max)
static __device__ __forceinline__ float vmax(float a, float b) {
  float r;
  asm("v_max_num_f32_e32 %0, %1, %2" : "=v"(r) : "v"(a), "v"(b));
  return r;
}

// 16-lane row max: DPP butterfly folded directly into v_max (1 instr/step)
static __device__ __forceinline__ float rowmax16(float a0, float a1) {
  float t = vmax(a0, a1), r;
  asm("v_max_num_f32_dpp %0, %1, %1 quad_perm:[1,0,3,2] row_mask:0xf bank_mask:0xf bound_ctrl:1"
      : "=v"(r) : "v"(t)); t = r;                       // xor 1
  asm("v_max_num_f32_dpp %0, %1, %1 quad_perm:[2,3,0,1] row_mask:0xf bank_mask:0xf bound_ctrl:1"
      : "=v"(r) : "v"(t)); t = r;                       // xor 2
  asm("v_max_num_f32_dpp %0, %1, %1 row_half_mirror row_mask:0xf bank_mask:0xf bound_ctrl:1"
      : "=v"(r) : "v"(t)); t = r;                       // combine quads within 8
  asm("v_max_num_f32_dpp %0, %1, %1 row_mirror row_mask:0xf bank_mask:0xf bound_ctrl:1"
      : "=v"(r) : "v"(t)); t = r;                       // combine halves within 16
  return t;
}

__global__ __launch_bounds__(256)
void mha_flash_bf16_kernel(const float* __restrict__ Q, const float* __restrict__ K,
                           const float* __restrict__ V, float* __restrict__ Out) {
  __shared__ __bf16 sK [64 * KP];        // K tile:  [key][feat]
  __shared__ __bf16 sVt[64 * KP];        // V tile transposed: [feat][key]
  __shared__ __bf16 sP [8][16 * 32];     // per-wave P scratch (16 rows x 32 keys)

  const int tid  = threadIdx.x;
  const int lane = tid & 31;
  const int wv   = tid >> 5;        // wave id 0..7
  const int n    = lane & 15;       // column / M-row index within fragment
  const int hl   = lane >> 4;       // half-wave select

  const int QT = T / 128;
  const int qt = blockIdx.x % QT;
  const int bh = blockIdx.x / QT;
  const int h  = bh % H;
  const int b  = bh / H;

  const int qbase = qt * 128 + wv * 16;          // this wave's 16 Q rows

  // ---- load Q A-fragments once (16x32 bf16 x2), pre-scaled by 0.125*log2(e) ----
  const float* qp = Q + ((size_t)(b * T + qbase + n) * U) + h * D;
  v16bf qa[2];
  {
    union { unsigned int w[8]; v16bf v; } u;
#pragma unroll
    for (int c = 0; c < 2; ++c) {
#pragma unroll
      for (int p = 0; p < 8; ++p) {
        const int kb = c * 32 + hl * 16 + p * 2;
        f32x2 f = *(const f32x2*)(qp + kb);
        u.w[p] = pack_bf2(f.x * QSC, f.y * QSC);
      }
      qa[c] = u.v;
    }
  }

  // all-ones B fragment: row-sum of P via the matrix core (softmax denominator)
  v16bf ones;
  {
    union { unsigned int w[8]; v16bf v; } u;
#pragma unroll
    for (int i = 0; i < 8; ++i) u.w[i] = 0x3F803F80u;   // bf16 1.0 x2
    ones = u.v;
  }

  float mrow[8];
#pragma unroll
  for (int r = 0; r < 8; ++r) mrow[r] = -__builtin_inff();
  v8f o0 = {}, o1 = {}, o2 = {}, o3 = {};
  v8f ol = {};                                   // running row-sums (l) per row

  const int srow = tid >> 2;        // staging: key row 0..63
  const int sqd  = tid & 3;         // staging: 16-feature quarter

  for (int kt = 0; kt < T; kt += 64) {
    __syncthreads();                // previous tile fully consumed
    // ---- stage K tile (64 keys x 64 feats) row-major bf16 ----
    {
      const float* kp = K + ((size_t)(b * T + kt + srow) * U) + h * D + sqd * 16;
      f32x4 f0 = *(const f32x4*)(kp + 0);
      f32x4 f1 = *(const f32x4*)(kp + 4);
      f32x4 f2 = *(const f32x4*)(kp + 8);
      f32x4 f3 = *(const f32x4*)(kp + 12);
      union { unsigned int w[8]; u32x4 q[2]; } pk;
      pk.w[0] = pack_bf2(f0.x, f0.y); pk.w[1] = pack_bf2(f0.z, f0.w);
      pk.w[2] = pack_bf2(f1.x, f1.y); pk.w[3] = pack_bf2(f1.z, f1.w);
      pk.w[4] = pack_bf2(f2.x, f2.y); pk.w[5] = pack_bf2(f2.z, f2.w);
      pk.w[6] = pack_bf2(f3.x, f3.y); pk.w[7] = pack_bf2(f3.z, f3.w);
      __bf16* d = &sK[srow * KP + sqd * 16];
      *(u32x4*)(d)     = pk.q[0];
      *(u32x4*)(d + 8) = pk.q[1];
      if (kt + 64 < T) __builtin_prefetch(kp + 64 * U, 0, 1);
    }
    // ---- stage V tile transposed: sVt[feat][key] bf16 ----
    {
      const float* vp = V + ((size_t)(b * T + kt + srow) * U) + h * D + sqd * 16;
      f32x4 f0 = *(const f32x4*)(vp + 0);
      f32x4 f1 = *(const f32x4*)(vp + 4);
      f32x4 f2 = *(const f32x4*)(vp + 8);
      f32x4 f3 = *(const f32x4*)(vp + 12);
      float fv[16] = { f0.x,f0.y,f0.z,f0.w, f1.x,f1.y,f1.z,f1.w,
                       f2.x,f2.y,f2.z,f2.w, f3.x,f3.y,f3.z,f3.w };
#pragma unroll
      for (int i = 0; i < 16; ++i)
        sVt[(sqd * 16 + i) * KP + srow] = (__bf16)fv[i];
      if (kt + 64 < T) __builtin_prefetch(vp + 64 * U, 0, 1);
    }
    __syncthreads();

    // ---- two 32-key subtiles per staged tile ----
#pragma unroll
    for (int ks = 0; ks < 64; ks += 32) {
      // S = Q x K^T  (two 16-key column groups); scores already in exp2 units
      v8f s0 = {}, s1 = {};
      s0 = wmma_bf16(qa[0], load_bf16x16(&sK[(ks + n)      * KP +      hl * 16]), s0);
      s0 = wmma_bf16(qa[1], load_bf16x16(&sK[(ks + n)      * KP + 32 + hl * 16]), s0);
      s1 = wmma_bf16(qa[0], load_bf16x16(&sK[(ks + 16 + n) * KP +      hl * 16]), s1);
      s1 = wmma_bf16(qa[1], load_bf16x16(&sK[(ks + 16 + n) * KP + 32 + hl * 16]), s1);

      // online softmax across the 32 columns (rows in VGPR index, cols across lanes)
      __bf16* pw = sP[wv];
#pragma unroll
      for (int r = 0; r < 8; ++r) {
        const float a0 = s0[r];
        const float a1 = s1[r];
        const float t  = rowmax16(a0, a1);                 // DPP butterfly, 16-lane row
        const float mn = vmax(mrow[r], t);
        const float al = __builtin_amdgcn_exp2f(mrow[r] - mn);
        const float p0 = __builtin_amdgcn_exp2f(a0 - mn);
        const float p1 = __builtin_amdgcn_exp2f(a1 - mn);
        mrow[r] = mn;
        o0[r] *= al; o1[r] *= al; o2[r] *= al; o3[r] *= al;
        ol[r] *= al;                                       // rescale running l too
        const int row = r + hl * 8;
        pw[row * 32 + n]      = (__bf16)p0;
        pw[row * 32 + 16 + n] = (__bf16)p1;
      }
      asm volatile("s_wait_dscnt 0x0" ::: "memory");       // P visible to own wave

      // reload P as an A-fragment (16x32 bf16): dwords map 1:1 onto elements 2p,2p+1
      v16bf pa;
      {
        union { unsigned int w[8]; v16bf v; } u;
#pragma unroll
        for (int p = 0; p < 8; ++p) {
          const int kb = hl * 16 + p * 2;
          u.w[p] = *(const unsigned int*)(pw + n * 32 + kb);
        }
        pa = u.v;
      }

      // O += P (16x32) x V (32x64); l += P x ones (row sums on the matrix core)
      o0 = wmma_bf16(pa, load_bf16x16(&sVt[(0  + n) * KP + ks + hl * 16]), o0);
      o1 = wmma_bf16(pa, load_bf16x16(&sVt[(16 + n) * KP + ks + hl * 16]), o1);
      o2 = wmma_bf16(pa, load_bf16x16(&sVt[(32 + n) * KP + ks + hl * 16]), o2);
      o3 = wmma_bf16(pa, load_bf16x16(&sVt[(48 + n) * KP + ks + hl * 16]), o3);
      ol = wmma_bf16(pa, ones, ol);
    }
  }

  // ---- epilogue: normalize by l and store (coalesced 64B row segments) ----
#pragma unroll
  for (int r = 0; r < 8; ++r) {
    const float inv = 1.0f / ol[r];
    const int row = qbase + r + hl * 8;
    float* op = Out + ((size_t)(b * T + row) * U) + h * D + n;
    op[0]  = o0[r] * inv;
    op[16] = o1[r] * inv;
    op[32] = o2[r] * inv;
    op[48] = o3[r] * inv;
  }
}

extern "C" void kernel_launch(void* const* d_in, const int* in_sizes, int n_in,
                              void* d_out, int out_size, void* d_ws, size_t ws_size,
                              hipStream_t stream) {
  (void)in_sizes; (void)n_in; (void)out_size; (void)d_ws; (void)ws_size;
  const float* q = (const float*)d_in[0];
  const float* k = (const float*)d_in[1];
  const float* v = (const float*)d_in[2];
  float* out = (float*)d_out;
  dim3 grid(BATCH * H * (T / 128));   // 1024 blocks
  dim3 block(256);                    // 8 wave32
  mha_flash_bf16_kernel<<<grid, block, 0, stream>>>(q, k, v, out);
}
